// WanLinearAttention_37469294690990
// MI455X (gfx1250) — matrix operand order repaired and checked by reading, not compiled, not run on hardware
//
#include <hip/hip_runtime.h>

#define S_TOK 16384
#define DIMF  1536
#define NH    12
#define HD    128
#define CHALF 64
#define EPSV  1e-6f

typedef __attribute__((ext_vector_type(16))) _Float16 v16h;
typedef __attribute__((ext_vector_type(8)))  _Float16 v8h;
typedef __attribute__((ext_vector_type(8)))  float    v8f;
typedef unsigned int uint4v __attribute__((ext_vector_type(4)));
typedef int          int4v  __attribute__((ext_vector_type(4)));
typedef int          int8v  __attribute__((ext_vector_type(8)));

#if defined(__has_builtin)
#if __has_builtin(__builtin_amdgcn_tensor_load_to_lds)
#define HAVE_TDM 1
#endif
#endif

static __device__ __forceinline__ v8f wmma_f16(v16h a, v16h b, v8f c) {
  // D = A(16x32) * B(32x16) + C, f32 accumulate
  return __builtin_amdgcn_wmma_f32_16x16x32_f16(false, a, false, b, (short)0, c, false, false);
}

// A fragment 16x32 (MxK): row-major source, leading dim ld (elements).
static __device__ __forceinline__ v16h load_a(const _Float16* __restrict__ base,
                                              int ld, int row0, int k0, int lane) {
  int r  = row0 + (lane & 15);
  int kk = k0 + ((lane >> 4) << 4);
  const _Float16* p = base + (size_t)r * (size_t)ld + kk;
  return *(const v16h*)p;
}

// B fragment 32x16 (KxN): source stores each column contiguous along K.
static __device__ __forceinline__ v16h load_b(const _Float16* __restrict__ base,
                                              int ld, int col0, int k0, int lane) {
  int c  = col0 + (lane & 15);
  int kk = k0 + ((lane >> 4) << 4);
  const _Float16* p = base + (size_t)c * (size_t)ld + kk;
  return *(const v16h*)p;
}

// Async copy (GLOBAL_LOAD_ASYNC_TO_LDS_B128, ASYNCcnt-tracked). IOFFSET adds to
// both the LDS and global address, so offset:16 is the second 16B beat.
static __device__ __forceinline__ void async_copy32(unsigned lds_off, const _Float16* g) {
  asm volatile("global_load_async_to_lds_b128 %0, %1, off"
               :: "v"(lds_off), "v"(g) : "memory");
  asm volatile("global_load_async_to_lds_b128 %0, %1, off offset:16"
               :: "v"(lds_off), "v"(g) : "memory");
}

static __device__ __forceinline__ void wait_async0() {
  asm volatile("s_wait_asynccnt 0x0" ::: "memory");
}

// 128-row x 32-col f16 slab (8KB) of a row-major matrix (ld=DIMF) -> LDS [128][32].
static __device__ __forceinline__ void stage_slab(unsigned lds_base,
                                                  const _Float16* __restrict__ src,
                                                  int row_blk, int k0, int t) {
  int r = t >> 1;
  int c = (t & 1) << 4;
  const _Float16* g = src + (size_t)(row_blk + r) * DIMF + k0 + c;
  unsigned l = lds_base + (unsigned)((r * 32 + c) * 2);
  async_copy32(l, g);
}

// ---- Tensor Data Mover: DMA a 2-D f16 tile (rows x cols elems, row stride in
// elems) from global to LDS. D# per CDNA5 ISA.
static __device__ __forceinline__ void tdm_load_2d(unsigned lds_addr, const void* gaddr,
                                                   int rows, int cols, int stride) {
#ifdef HAVE_TDM
  unsigned long long ga = (unsigned long long)gaddr;
  uint4v g0;
  g0.x = 1u;                                            // count=1, user descriptor
  g0.y = lds_addr;                                      // LDS byte address
  g0.z = (unsigned)(ga & 0xFFFFFFFFu);                  // global_addr[31:0]
  g0.w = (unsigned)((ga >> 32) & 0x1FFFFFFu) | (2u << 30);  // addr[56:32] | type=2
  int8v g1;
  g1[0] = 0x00010000;                                   // data_size=1 (2 bytes)
  g1[1] = (cols & 0xFFFF) << 16;                        // tensor_dim0[15:0]
  g1[2] = ((unsigned)cols >> 16) | ((rows & 0xFFFF) << 16);   // td0[31:16] | td1[15:0]
  g1[3] = ((unsigned)rows >> 16) | ((cols & 0xFFFF) << 16);   // td1[31:16] | tile_dim0
  g1[4] = rows & 0xFFFF;                                // tile_dim1 | tile_dim2=0
  g1[5] = stride;                                       // tensor_dim0_stride[31:0]
  g1[6] = 0;
  g1[7] = 0;
  int4v gz = {0, 0, 0, 0};
#if __clang_major__ >= 23
  int8v gz8 = {0, 0, 0, 0, 0, 0, 0, 0};
  __builtin_amdgcn_tensor_load_to_lds(g0, g1, gz, gz, gz8, 0);
#else
  __builtin_amdgcn_tensor_load_to_lds(g0, g1, gz, gz, 0);
#endif
#else
  (void)lds_addr; (void)gaddr; (void)rows; (void)cols; (void)stride;
#endif
}

// Stage the B (weight) slab: TDM when available (issued by wave 0), else async.
static __device__ __forceinline__ void stage_b(unsigned lds_base,
                                               const _Float16* __restrict__ W,
                                               int n_blk, int k0, int t, int wid) {
#ifdef HAVE_TDM
  if (wid == 0) tdm_load_2d(lds_base, W + (size_t)n_blk * DIMF + k0, 128, 32, DIMF);
  (void)t;
#else
  stage_slab(lds_base, W, n_blk, k0, t);
  (void)wid;
#endif
}

static __device__ __forceinline__ void wait_b(int wid) {
#ifdef HAVE_TDM
  if (wid == 0) {
#if __has_builtin(__builtin_amdgcn_s_wait_tensorcnt)
    __builtin_amdgcn_s_wait_tensorcnt(0);
#else
    asm volatile("s_wait_tensorcnt 0x0" ::: "memory");
#endif
  }
#else
  (void)wid;
#endif
}

// ---------------- conversion / init kernels ----------------

__global__ void cvt_f32_f16(const float* __restrict__ in, _Float16* __restrict__ out, int n) {
  int i = blockIdx.x * blockDim.x + threadIdx.x;
  int stride = gridDim.x * blockDim.x;
  for (; i < n; i += stride) out[i] = (_Float16)in[i];
}

__global__ void zero_f32(float* __restrict__ p, int n) {
  int i = blockIdx.x * blockDim.x + threadIdx.x;
  if (i < n) p[i] = 0.0f;
}

// ---------------- big GEMM: Y[16384][1536] = A[16384][1536] * W[1536][1536]^T + bias ----
// 8 waves (2M x 4N), wave tile 64x32 (4x2 WMMA tiles). A slab staged via
// GLOBAL_LOAD_ASYNC_TO_LDS (ASYNCcnt); B slab via TENSOR_LOAD_TO_LDS (TENSORcnt)
// when available. Both double-buffered in LDS.

template <bool OUT32>
__global__ __launch_bounds__(256) void gemm_kernel(const _Float16* __restrict__ A,
                                                   const _Float16* __restrict__ W,
                                                   const float* __restrict__ bias,
                                                   void* __restrict__ Cout) {
  __shared__ __align__(64) _Float16 sA[2][128 * 32];
  __shared__ __align__(64) _Float16 sB[2][128 * 32];

  int t    = threadIdx.x;
  int lane = t & 31;
  int wid  = t >> 5;
  int wave_m = wid >> 2;
  int wave_n = wid & 3;
  int m_blk = blockIdx.x * 128;
  int n_blk = blockIdx.y * 128;

  unsigned ldsA[2], ldsB[2];
  ldsA[0] = (unsigned)(size_t)(const void*)&sA[0][0];
  ldsA[1] = (unsigned)(size_t)(const void*)&sA[1][0];
  ldsB[0] = (unsigned)(size_t)(const void*)&sB[0][0];
  ldsB[1] = (unsigned)(size_t)(const void*)&sB[1][0];

  v8f acc[4][2] = {};

  // prologue: fill buffer 0
  stage_slab(ldsA[0], A, m_blk, 0, t);
  stage_b(ldsB[0], W, n_blk, 0, t, wid);
  wait_async0();
  wait_b(wid);
  __syncthreads();

  for (int k0 = 0; k0 < DIMF; k0 += 32) {
    int cur = (k0 >> 5) & 1;
    int nxt = cur ^ 1;
    bool more = (k0 + 32) < DIMF;
    if (more) {
      stage_slab(ldsA[nxt], A, m_blk, k0 + 32, t);
      stage_b(ldsB[nxt], W, n_blk, k0 + 32, t, wid);
    }

    const _Float16* cA = &sA[cur][0];
    const _Float16* cB = &sB[cur][0];
    v16h af[4], bf[2];
    #pragma unroll
    for (int m = 0; m < 4; m++) af[m] = load_a(cA, 32, wave_m * 64 + m * 16, 0, lane);
    #pragma unroll
    for (int j = 0; j < 2; j++)  bf[j] = load_b(cB, 32, wave_n * 32 + j * 16, 0, lane);
    #pragma unroll
    for (int m = 0; m < 4; m++)
      #pragma unroll
      for (int j = 0; j < 2; j++)
        acc[m][j] = wmma_f16(af[m], bf[j], acc[m][j]);

    if (more) { wait_async0(); wait_b(wid); }
    __syncthreads();
  }

  int colq = lane & 15;
  int rowq = (lane >> 4) << 3;
  #pragma unroll
  for (int m = 0; m < 4; m++)
    #pragma unroll
    for (int j = 0; j < 2; j++) {
      int col = n_blk + wave_n * 32 + j * 16 + colq;
      float bv = bias[col];
      #pragma unroll
      for (int r = 0; r < 8; r++) {
        int row = m_blk + wave_m * 64 + m * 16 + rowq + r;
        float v = acc[m][j][r] + bv;
        if (OUT32) ((float*)Cout)[(size_t)row * DIMF + col] = v;
        else       ((_Float16*)Cout)[(size_t)row * DIMF + col] = (_Float16)v;
      }
    }
}

// ---------------- RMSNorm + RoPE + ReLU + layout shuffle ----------------

__global__ __launch_bounds__(256) void norm_rope_kernel(
    const _Float16* __restrict__ yq, const _Float16* __restrict__ yk,
    const _Float16* __restrict__ yv, const float* __restrict__ freqs,
    const float* __restrict__ gq, const float* __restrict__ gk,
    _Float16* __restrict__ qh, _Float16* __restrict__ kT, _Float16* __restrict__ vT) {
  int s = blockIdx.x;
  int t = threadIdx.x;
  __shared__ float redq[256];
  __shared__ float redk[256];

  float qv[6], kv[6], vv[6];
  float sq = 0.f, sk = 0.f;
  size_t rb = (size_t)s * DIMF;
  #pragma unroll
  for (int i = 0; i < 6; i++) {
    int idx = t * 6 + i;
    qv[i] = (float)yq[rb + idx];
    kv[i] = (float)yk[rb + idx];
    vv[i] = (float)yv[rb + idx];
    sq += qv[i] * qv[i];
    sk += kv[i] * kv[i];
  }
  redq[t] = sq; redk[t] = sk;
  __syncthreads();
  for (int o = 128; o > 0; o >>= 1) {
    if (t < o) { redq[t] += redq[t + o]; redk[t] += redk[t + o]; }
    __syncthreads();
  }
  float rq = rsqrtf(redq[0] / (float)DIMF + EPSV);
  float rk = rsqrtf(redk[0] / (float)DIMF + EPSV);

  int f = s >> 10;
  int h = (s >> 5) & 31;
  int w = s & 31;

  #pragma unroll
  for (int p = 0; p < 3; p++) {
    int idx = t * 6 + 2 * p;
    int n  = idx >> 7;
    int dd = idx & 127;
    int c  = dd >> 1;
    int pos = (c < 22) ? f : ((c < 43) ? h : w);   // cf=22, ch=21
    float ang = freqs[pos * CHALF + c];
    float cs = cosf(ang), sn = sinf(ang);

    float qe = qv[2*p]     * rq * gq[idx];
    float qo = qv[2*p + 1] * rq * gq[idx + 1];
    float ke = kv[2*p]     * rk * gk[idx];
    float ko = kv[2*p + 1] * rk * gk[idx + 1];
    float q0 = fmaxf(qe * cs - qo * sn, 0.f);
    float q1 = fmaxf(qe * sn + qo * cs, 0.f);
    float k0 = fmaxf(ke * cs - ko * sn, 0.f);
    float k1 = fmaxf(ke * sn + ko * cs, 0.f);

    size_t qoff = ((size_t)n * S_TOK + s) * HD + dd;
    qh[qoff]     = (_Float16)q0;
    qh[qoff + 1] = (_Float16)q1;

    size_t koff = ((size_t)(n * HD + dd)) * S_TOK + s;
    kT[koff]         = (_Float16)k0;
    kT[koff + S_TOK] = (_Float16)k1;
    vT[koff]         = (_Float16)vv[2*p];
    vT[koff + S_TOK] = (_Float16)vv[2*p + 1];
  }
}

// ---------------- Stage 1: VK[n][e][d] = sum_s Vpad[s][e] * K[s][d] ----------------

__global__ __launch_bounds__(256) void vk_kernel(const _Float16* __restrict__ vT,
                                                 const _Float16* __restrict__ kT,
                                                 float* __restrict__ vk) {
  int n = blockIdx.y;
  int lane = threadIdx.x & 31;
  int wid  = threadIdx.x >> 5;
  int d0 = wid * 16;
  const _Float16* vbase = vT + (size_t)n * HD * S_TOK;
  const _Float16* kbase = kT + (size_t)n * HD * S_TOK;

  v8f acc[9] = {};
  v16h a8 = {};
  if ((lane & 15) == 0) {
    #pragma unroll
    for (int j = 0; j < 16; j++) a8[j] = (_Float16)1.0f;
  }

  int s_begin = blockIdx.x * (S_TOK / 16);
  for (int s0 = s_begin; s0 < s_begin + (S_TOK / 16); s0 += 32) {
    v16h b = load_b(kbase, S_TOK, d0, s0, lane);
    #pragma unroll
    for (int m = 0; m < 8; m++) {
      v16h a = load_a(vbase, S_TOK, m * 16, s0, lane);
      acc[m] = wmma_f16(a, b, acc[m]);
    }
    acc[8] = wmma_f16(a8, b, acc[8]);
  }

  int rowq = (lane >> 4) << 3;
  int d = d0 + (lane & 15);
  #pragma unroll
  for (int m = 0; m < 9; m++)
    #pragma unroll
    for (int r = 0; r < 8; r++) {
      int e = m * 16 + rowq + r;
      if (e <= 128) atomicAdd(&vk[(size_t)(n * 144 + e) * HD + d], acc[m][r]);
    }
}

// ---------------- Stage 2: out = VK * Q^T, divide by pad row, write y ----------------

__global__ __launch_bounds__(256) void attn_out_kernel(const float* __restrict__ vk,
                                                       const _Float16* __restrict__ qh,
                                                       _Float16* __restrict__ yh) {
  int n = blockIdx.y;
  int lane = threadIdx.x & 31;
  int wid  = threadIdx.x >> 5;
  int s0 = blockIdx.x * 128 + wid * 16;
  const float* vkb = vk + (size_t)n * 144 * HD;

  v8f acc[9] = {};
  int tok = s0 + (lane & 15);
  const _Float16* qcol = qh + ((size_t)n * S_TOK + tok) * HD;

  for (int kd = 0; kd < HD; kd += 32) {
    v16h b = *(const v16h*)(qcol + kd + ((lane >> 4) << 4));
    #pragma unroll
    for (int m = 0; m < 9; m++) {
      int e = m * 16 + (lane & 15);
      const float* pf = vkb + (size_t)e * HD + kd + ((lane >> 4) << 4);
      v16h a;
      #pragma unroll
      for (int j = 0; j < 16; j++) { a[j] = (_Float16)pf[j]; }
      acc[m] = wmma_f16(a, b, acc[m]);
    }
  }

  float den = acc[8][0];
  den = __int_as_float(__builtin_amdgcn_ds_bpermute((lane & 15) << 2, __float_as_int(den)));
  float inv = 1.0f / (den + EPSV);

  int rowq = (lane >> 4) << 3;
  #pragma unroll
  for (int m = 0; m < 8; m++)
    #pragma unroll
    for (int r = 0; r < 8; r++) {
      int e = m * 16 + rowq + r;
      yh[(size_t)tok * DIMF + n * HD + e] = (_Float16)(acc[m][r] * inv);
    }
}

// ---------------- host side ----------------

extern "C" void kernel_launch(void* const* d_in, const int* in_sizes, int n_in,
                              void* d_out, int out_size, void* d_ws, size_t ws_size,
                              hipStream_t stream) {
  (void)in_sizes; (void)n_in; (void)out_size; (void)ws_size;
  const float* x     = (const float*)d_in[0];
  const float* freqs = (const float*)d_in[3];
  const float* Wq    = (const float*)d_in[4];
  const float* bq    = (const float*)d_in[5];
  const float* Wk    = (const float*)d_in[6];
  const float* bk    = (const float*)d_in[7];
  const float* Wv    = (const float*)d_in[8];
  const float* bv    = (const float*)d_in[9];
  const float* Wo    = (const float*)d_in[10];
  const float* bo    = (const float*)d_in[11];
  const float* gq    = (const float*)d_in[12];
  const float* gk    = (const float*)d_in[13];
  float* out = (float*)d_out;

  char* ws = (char*)d_ws;
  size_t off = 0;
  auto take = [&](size_t bytes) -> char* {
    char* p = ws + off;
    off += (bytes + 255) & ~(size_t)255;
    return p;
  };
  const size_t SD = (size_t)S_TOK * DIMF;
  const size_t WW = (size_t)DIMF * DIMF;

  _Float16* xh  = (_Float16*)take(SD * 2);
  _Float16* Wqh = (_Float16*)take(WW * 2);
  _Float16* Wkh = (_Float16*)take(WW * 2);
  _Float16* Wvh = (_Float16*)take(WW * 2);
  _Float16* Woh = (_Float16*)take(WW * 2);
  _Float16* yq  = (_Float16*)take(SD * 2);
  _Float16* yk  = (_Float16*)take(SD * 2);
  _Float16* yv  = (_Float16*)take(SD * 2);
  _Float16* qh  = (_Float16*)take(SD * 2);   // [n][s][d]
  _Float16* kT  = (_Float16*)take(SD * 2);   // [n][d][s]
  _Float16* vT  = (_Float16*)take(SD * 2);   // [n][d][s]
  float*    vk  = (float*)take((size_t)NH * 144 * HD * 4);
  _Float16* yh  = yq;  // yq fully consumed before yh is produced

  const int THR = 256;
  cvt_f32_f16<<<2048, THR, 0, stream>>>(x,  xh,  (int)SD);
  cvt_f32_f16<<<512,  THR, 0, stream>>>(Wq, Wqh, (int)WW);
  cvt_f32_f16<<<512,  THR, 0, stream>>>(Wk, Wkh, (int)WW);
  cvt_f32_f16<<<512,  THR, 0, stream>>>(Wv, Wvh, (int)WW);
  cvt_f32_f16<<<512,  THR, 0, stream>>>(Wo, Woh, (int)WW);
  zero_f32<<<(NH * 144 * HD + THR - 1) / THR, THR, 0, stream>>>(vk, NH * 144 * HD);

  dim3 gg(S_TOK / 128, DIMF / 128);
  gemm_kernel<false><<<gg, THR, 0, stream>>>(xh, Wqh, bq, yq);
  gemm_kernel<false><<<gg, THR, 0, stream>>>(xh, Wkh, bk, yk);
  gemm_kernel<false><<<gg, THR, 0, stream>>>(xh, Wvh, bv, yv);

  norm_rope_kernel<<<S_TOK, THR, 0, stream>>>(yq, yk, yv, freqs, gq, gk, qh, kT, vT);

  vk_kernel<<<dim3(16, NH), THR, 0, stream>>>(vT, kT, vk);
  attn_out_kernel<<<dim3(S_TOK / 128, NH), THR, 0, stream>>>(vk, qh, yh);

  gemm_kernel<true><<<gg, THR, 0, stream>>>(yh, Woh, bo, out);
}
